// MultiHeadSelfAttention_22359599743303
// MI455X (gfx1250) — compile-verified, matrix operand order checked by reading, not run on hardware
//
#include <hip/hip_runtime.h>
#include <hip/hip_bf16.h>

// ---------------------------------------------------------------------------
// Problem constants (B=2, S=2048, E=1024, H=16, D=64)
// ---------------------------------------------------------------------------
#define BATCH 2
#define SEQ   2048
#define EMB   1024
#define HEADS 16
#define HD    64
#define E3    (3 * EMB)       // 3072
#define MROWS (BATCH * SEQ)   // 4096
#define QMAX  255.0f          // 2^8 - 1

typedef __attribute__((ext_vector_type(16))) __bf16 v16bf;
typedef __attribute__((ext_vector_type(8)))  __bf16 v8bf;
typedef __attribute__((ext_vector_type(8)))  float  v8f;

__device__ __forceinline__ v8f wmma_bf16(v16bf a, v16bf b, v8f c) {
    return __builtin_amdgcn_wmma_f32_16x16x32_bf16(
        false, a, false, b, (short)0, c, false, false);
}

// CDNA5 async copy: global memory -> LDS, 16 bytes per active lane.
// Low 32 bits of a generic LDS pointer are the LDS byte address (aperture
// truncation per ISA 10.2).  Tracked by ASYNCcnt.
__device__ __forceinline__ void async_copy_b128(void* lds_dst, const void* gsrc) {
    unsigned l = (unsigned)(unsigned long long)lds_dst;
    asm volatile("global_load_async_to_lds_b128 %0, %1, off"
                 :: "v"(l), "v"(gsrc) : "memory");
}
__device__ __forceinline__ void wait_async() {
    asm volatile("s_wait_asynccnt 0" ::: "memory");
}

// xor-butterfly within each 16-lane row via v_permlane16_b32 (pure VALU,
// replaces ds_bpermute + s_wait_dscnt round trips).
__device__ __forceinline__ float pl16(float v, unsigned lo, unsigned hi) {
    return __int_as_float(__builtin_amdgcn_permlane16(
        __float_as_int(v), __float_as_int(v), lo, hi, false, false));
}
__device__ __forceinline__ float redmax16(float v) {
    v = fmaxf(v, pl16(v, 0x67452301u, 0xEFCDAB89u));   // xor 1
    v = fmaxf(v, pl16(v, 0x54761032u, 0xDCFE98BAu));   // xor 2
    v = fmaxf(v, pl16(v, 0x32107654u, 0xBA98FEDCu));   // xor 4
    v = fmaxf(v, pl16(v, 0xFEDCBA98u, 0x76543210u));   // xor 8
    return v;
}
__device__ __forceinline__ float redsum16(float v) {
    v += pl16(v, 0x67452301u, 0xEFCDAB89u);
    v += pl16(v, 0x54761032u, 0xDCFE98BAu);
    v += pl16(v, 0x32107654u, 0xBA98FEDCu);
    v += pl16(v, 0xFEDCBA98u, 0x76543210u);
    return v;
}

// Load one 16x32 bf16 fragment in the CDNA5 A/B register layout from a
// row-major LDS tile.  Per ISA 7.12.2: lane L holds row (L&15);
// packed K = 16*group + 8*(L>>4) + j.
__device__ __forceinline__ v16bf load_frag(const __bf16* rowbase, int stride,
                                           int m15, int hf) {
    const __bf16* p = rowbase + m15 * stride + hf * 8;
    union { v16bf v; v8bf h[2]; } u;
    u.h[0] = *(const v8bf*)(p);
    u.h[1] = *(const v8bf*)(p + 16);
    return u.v;
}

// ---------------------------------------------------------------------------
// 1) min/max reduction for fake-quantization
// ---------------------------------------------------------------------------
__global__ void minmax_partial_kernel(const float* __restrict__ w, long n,
                                      float* __restrict__ partial) {
    __shared__ float smin[256];
    __shared__ float smax[256];
    float mn =  3.402823466e38f;
    float mx = -3.402823466e38f;
    long stride = (long)gridDim.x * blockDim.x;
    for (long i = (long)blockIdx.x * blockDim.x + threadIdx.x; i < n; i += stride) {
        float v = w[i];
        mn = fminf(mn, v);
        mx = fmaxf(mx, v);
    }
    smin[threadIdx.x] = mn;
    smax[threadIdx.x] = mx;
    __syncthreads();
    for (int s = 128; s > 0; s >>= 1) {
        if (threadIdx.x < s) {
            smin[threadIdx.x] = fminf(smin[threadIdx.x], smin[threadIdx.x + s]);
            smax[threadIdx.x] = fmaxf(smax[threadIdx.x], smax[threadIdx.x + s]);
        }
        __syncthreads();
    }
    if (threadIdx.x == 0) {
        partial[2 * blockIdx.x + 0] = smin[0];
        partial[2 * blockIdx.x + 1] = smax[0];
    }
}

__global__ void minmax_final_kernel(const float* __restrict__ partial, int nblk,
                                    float* __restrict__ out2) {
    __shared__ float smin[256];
    __shared__ float smax[256];
    float mn =  3.402823466e38f;
    float mx = -3.402823466e38f;
    for (int i = threadIdx.x; i < nblk; i += blockDim.x) {
        mn = fminf(mn, partial[2 * i + 0]);
        mx = fmaxf(mx, partial[2 * i + 1]);
    }
    smin[threadIdx.x] = mn;
    smax[threadIdx.x] = mx;
    __syncthreads();
    for (int s = 128; s > 0; s >>= 1) {
        if (threadIdx.x < s) {
            smin[threadIdx.x] = fminf(smin[threadIdx.x], smin[threadIdx.x + s]);
            smax[threadIdx.x] = fmaxf(smax[threadIdx.x], smax[threadIdx.x + s]);
        }
        __syncthreads();
    }
    if (threadIdx.x == 0) {
        float xmin = smin[0];
        float xmax = smax[0];
        out2[0] = xmin;
        out2[1] = (xmax - xmin) / (QMAX + 1e-8f);
    }
}

// ---------------------------------------------------------------------------
// 2) fake-quantize weights -> bf16 ; f32 -> bf16 conversion
// ---------------------------------------------------------------------------
__global__ void quantize_bf16_kernel(const float* __restrict__ w,
                                     __bf16* __restrict__ out, long n,
                                     const float* __restrict__ p2) {
    float xmin = p2[0];
    float scale = p2[1];
    long stride = (long)gridDim.x * blockDim.x;
    for (long i = (long)blockIdx.x * blockDim.x + threadIdx.x; i < n; i += stride) {
        float q = roundf(fminf(fmaxf((w[i] - xmin) / (scale + 1e-8f), 0.0f), QMAX));
        out[i] = (__bf16)(q * scale + xmin);
    }
}

__global__ void cvt_bf16_kernel(const float* __restrict__ x,
                                __bf16* __restrict__ out, long n) {
    long stride = (long)gridDim.x * blockDim.x;
    for (long i = (long)blockIdx.x * blockDim.x + threadIdx.x; i < n; i += stride)
        out[i] = (__bf16)x[i];
}

// ---------------------------------------------------------------------------
// 3) Tiled bf16 WMMA GEMM:  Out[M,N] = A[M,K] * B[K,N]   (row-major)
//    128x128 block tile, 8 waves, wave tile 32x64, K-step 32.
//    A tile staged with global_load_async_to_lds_b128.
// ---------------------------------------------------------------------------
#define GPAD 48   // LDS row stride in elements (96 B, 16B-aligned)

template <bool OUT_BF16>
__global__ __launch_bounds__(256)
void gemm_bf16_kernel(const __bf16* __restrict__ A, const __bf16* __restrict__ Bm,
                      void* __restrict__ Out, int M, int N, int K) {
    __shared__ __bf16 lA[128 * GPAD];   // [row][k]
    __shared__ __bf16 lBt[128 * GPAD];  // [n][k]  (B transposed)

    const int tid  = threadIdx.x;
    const int lane = tid & 31;
    const int wave = tid >> 5;
    const int m15  = lane & 15;
    const int hf   = lane >> 4;
    const int wr   = wave >> 1;
    const int wc   = wave & 1;
    const int m0   = blockIdx.y * 128;
    const int n0   = blockIdx.x * 128;

    v8f acc[2][4] = {};

    const int ar = tid >> 1, ac = (tid & 1) * 16;       // A-tile: 128 x 32
    const int bk = tid >> 3, bn = (tid & 7) * 16;       // B-tile: 32 x 128

    for (int k0 = 0; k0 < K; k0 += 32) {
        // stage A via async copy (memory -> LDS, no VGPR round trip)
        {
            const __bf16* g = A + (long)(m0 + ar) * K + k0 + ac;
            __builtin_prefetch(g + 32, 0, 1);            // global_prefetch_b8
            async_copy_b128(&lA[ar * GPAD + ac],     g);
            async_copy_b128(&lA[ar * GPAD + ac + 8], g + 8);
        }
        // stage B transposed -> lBt[n][k] (must round-trip through VGPRs)
        {
            const __bf16* g = Bm + (long)(k0 + bk) * N + n0 + bn;
            v8bf b0 = *(const v8bf*)(g);
            v8bf b1 = *(const v8bf*)(g + 8);
#pragma unroll
            for (int i = 0; i < 8; ++i) lBt[(bn + i) * GPAD + bk] = b0[i];
#pragma unroll
            for (int i = 0; i < 8; ++i) lBt[(bn + 8 + i) * GPAD + bk] = b1[i];
        }
        wait_async();
        __syncthreads();

        v16bf af[2], bfr[4];
#pragma unroll
        for (int i = 0; i < 2; ++i)
            af[i] = load_frag(&lA[(wr * 32 + i * 16) * GPAD], GPAD, m15, hf);
#pragma unroll
        for (int j = 0; j < 4; ++j)
            bfr[j] = load_frag(&lBt[(wc * 64 + j * 16) * GPAD], GPAD, m15, hf);
#pragma unroll
        for (int i = 0; i < 2; ++i)
#pragma unroll
            for (int j = 0; j < 4; ++j)
                acc[i][j] = wmma_bf16(af[i], bfr[j], acc[i][j]);
        __syncthreads();
    }

#pragma unroll
    for (int i = 0; i < 2; ++i)
#pragma unroll
        for (int j = 0; j < 4; ++j)
#pragma unroll
            for (int r = 0; r < 8; ++r) {
                int row = m0 + wr * 32 + i * 16 + r + 8 * hf;
                int col = n0 + wc * 64 + j * 16 + m15;
                float v = acc[i][j][r];
                if (OUT_BF16)
                    ((__bf16*)Out)[(long)row * N + col] = (__bf16)v;
                else
                    ((float*)Out)[(long)row * N + col] = v;
            }
}

// ---------------------------------------------------------------------------
// 4) In-place RoPE on Q and K slices of qkv (layout [B*S, 3, H, D])
// ---------------------------------------------------------------------------
__global__ void rope_kernel(__bf16* __restrict__ qkv) {
    const int half = HD / 2; // 32
    long idx = (long)blockIdx.x * blockDim.x + threadIdx.x;
    const long total = (long)BATCH * SEQ * 2 * HEADS * half;
    if (idx >= total) return;
    int i = (int)(idx % half);  idx /= half;
    int h = (int)(idx % HEADS); idx /= HEADS;
    int which = (int)(idx % 2); idx /= 2;     // 0 = Q, 1 = K
    int s = (int)(idx % SEQ);
    int b = (int)(idx / SEQ);
    float freq = __powf(10000.0f, -(float)i / (float)half);
    float ang = (float)s * freq;
    float sn = __sinf(ang), cs = __cosf(ang);
    long base = ((long)(b * SEQ + s)) * E3 + (long)which * EMB + (long)h * HD;
    float x1 = (float)qkv[base + i];
    float x2 = (float)qkv[base + i + half];
    qkv[base + i]        = (__bf16)(x1 * cs - x2 * sn);
    qkv[base + i + half] = (__bf16)(x1 * sn + x2 * cs);
}

// ---------------------------------------------------------------------------
// 5) Flash attention: block = (128 query rows) x (one b,h); 8 waves x 16 rows.
//    Q/K tiles staged via async copy; V staged transposed through VGPRs.
// ---------------------------------------------------------------------------
#define APAD 72   // 144 B stride, 16B aligned

__global__ __launch_bounds__(256)
void flash_attn_kernel(const __bf16* __restrict__ qkv, __bf16* __restrict__ outbf) {
    __shared__ __bf16 lQ[128 * APAD];      // [row][d]
    __shared__ __bf16 lK[64 * APAD];       // [key][d]   == Bt for QK^T
    __shared__ __bf16 lVt[64 * APAD];      // [d][key]   == Bt for P*V
    __shared__ __bf16 lP[8 * 16 * APAD];   // per-wave P tile 16 x 64

    const int tid  = threadIdx.x;
    const int lane = tid & 31;
    const int wave = tid >> 5;
    const int m15  = lane & 15;
    const int hf   = lane >> 4;

    const int qt = blockIdx.x;
    const int bh = blockIdx.y;
    const int b = bh / HEADS, h = bh % HEADS;
    const int row0 = qt * 128;

    const long qbase = (long)b * SEQ * E3 + (long)h * HD;
    const long kbase = qbase + EMB;
    const long vbase = qbase + 2 * EMB;

    // stage Q tile (128 x 64) asynchronously
    {
        int r = tid >> 1, d0 = (tid & 1) * 32;
        const __bf16* g = qkv + qbase + (long)(row0 + r) * E3 + d0;
#pragma unroll
        for (int i = 0; i < 4; ++i)
            async_copy_b128(&lQ[r * APAD + d0 + i * 8], g + i * 8);
    }

    v8f o[4] = {};
    float mrow[8], lrow[8];
#pragma unroll
    for (int r = 0; r < 8; ++r) { mrow[r] = -3.402823466e38f; lrow[r] = 0.0f; }
    wait_async();
    __syncthreads();

    const int kvr = tid >> 2, kvd = (tid & 3) * 16;   // 64 x 64 tile staging

    for (int kt = 0; kt < SEQ / 64; ++kt) {
        // stage K tile row-major (async)
        {
            const __bf16* g = qkv + kbase + (long)(kt * 64 + kvr) * E3 + kvd;
            async_copy_b128(&lK[kvr * APAD + kvd],     g);
            async_copy_b128(&lK[kvr * APAD + kvd + 8], g + 8);
        }
        // stage V tile transposed (VGPR round trip)
        {
            const __bf16* g = qkv + vbase + (long)(kt * 64 + kvr) * E3 + kvd;
            v8bf v0 = *(const v8bf*)(g);
            v8bf v1 = *(const v8bf*)(g + 8);
#pragma unroll
            for (int i = 0; i < 8; ++i) lVt[(kvd + i) * APAD + kvr] = v0[i];
#pragma unroll
            for (int i = 0; i < 8; ++i) lVt[(kvd + 8 + i) * APAD + kvr] = v1[i];
        }
        wait_async();
        __syncthreads();

        // scores: S = Q(16x64) * K^T  -> 4 accumulators of 16x16
        v8f sacc[4] = {};
        v16bf qa0 = load_frag(&lQ[(wave * 16) * APAD + 0],  APAD, m15, hf);
        v16bf qa1 = load_frag(&lQ[(wave * 16) * APAD + 32], APAD, m15, hf);
#pragma unroll
        for (int j = 0; j < 4; ++j) {
            v16bf kb0 = load_frag(&lK[(j * 16) * APAD + 0],  APAD, m15, hf);
            v16bf kb1 = load_frag(&lK[(j * 16) * APAD + 32], APAD, m15, hf);
            sacc[j] = wmma_bf16(qa0, kb0, sacc[j]);
            sacc[j] = wmma_bf16(qa1, kb1, sacc[j]);
        }

        // online softmax: row r of acc = query row (r + 8*hf), spread over a
        // 16-lane half; reduce with v_permlane16_b32 butterflies.
#pragma unroll
        for (int r = 0; r < 8; ++r) {
            float rm = -3.402823466e38f;
#pragma unroll
            for (int j = 0; j < 4; ++j) {
                sacc[j][r] *= 0.125f;               // 1/sqrt(64)
                rm = fmaxf(rm, sacc[j][r]);
            }
            rm = redmax16(rm);
            float nm = fmaxf(mrow[r], rm);
            float alpha = __expf(mrow[r] - nm);
            float rs = 0.0f;
#pragma unroll
            for (int j = 0; j < 4; ++j) {
                float p = __expf(sacc[j][r] - nm);
                sacc[j][r] = p;
                rs += p;
            }
            rs = redsum16(rs);
            lrow[r] = lrow[r] * alpha + rs;
            mrow[r] = nm;
#pragma unroll
            for (int t = 0; t < 4; ++t) o[t][r] *= alpha;
        }

        // convert P from C-layout to A-layout via per-wave LDS round-trip
        __bf16* pw = &lP[wave * 16 * APAD];
#pragma unroll
        for (int j = 0; j < 4; ++j)
#pragma unroll
            for (int r = 0; r < 8; ++r)
                pw[(r + 8 * hf) * APAD + j * 16 + m15] = (__bf16)sacc[j][r];
        __syncthreads();

        // O += P(16x64) * V(64x64)
#pragma unroll
        for (int ds = 0; ds < 2; ++ds) {
            v16bf pa = load_frag(pw + ds * 32, APAD, m15, hf);
#pragma unroll
            for (int t = 0; t < 4; ++t) {
                v16bf vb = load_frag(&lVt[(t * 16) * APAD + ds * 32], APAD, m15, hf);
                o[t] = wmma_bf16(pa, vb, o[t]);
            }
        }
        __syncthreads();
    }

    // normalize and store: attn_out[(b*S + row), h*64 + d] as bf16
#pragma unroll
    for (int r = 0; r < 8; ++r) {
        float inv = lrow[r] > 0.0f ? 1.0f / lrow[r] : 0.0f;
        int row = row0 + wave * 16 + r + 8 * hf;
#pragma unroll
        for (int t = 0; t < 4; ++t)
            outbf[(long)(b * SEQ + row) * EMB + h * HD + t * 16 + m15] =
                (__bf16)(o[t][r] * inv);
    }
}

// ---------------------------------------------------------------------------
// Launch
// ---------------------------------------------------------------------------
extern "C" void kernel_launch(void* const* d_in, const int* in_sizes, int n_in,
                              void* d_out, int out_size, void* d_ws, size_t ws_size,
                              hipStream_t stream) {
    (void)in_sizes; (void)n_in; (void)out_size; (void)ws_size;
    const float* x     = (const float*)d_in[0];   // (B,S,E)
    const float* w_qkv = (const float*)d_in[1];   // (E, 3E)
    const float* w_out = (const float*)d_in[2];   // (E, E)
    float* out = (float*)d_out;                   // (B,S,E) f32

    char* ws = (char*)d_ws;
    float*  partA    = (float*)(ws + 0);
    float*  partB    = (float*)(ws + 1024);
    float*  paramsQ  = (float*)(ws + 2048);
    float*  paramsO  = (float*)(ws + 2048 + 64);
    size_t off = 4096;
    __bf16* x_bf     = (__bf16*)(ws + off); off += (size_t)MROWS * EMB * 2;
    __bf16* wqkv_bf  = (__bf16*)(ws + off); off += (size_t)EMB * E3 * 2;
    __bf16* wout_bf  = (__bf16*)(ws + off); off += (size_t)EMB * EMB * 2;
    __bf16* qkv_bf   = (__bf16*)(ws + off); off += (size_t)MROWS * E3 * 2;
    __bf16* attn_bf  = (__bf16*)(ws + off);

    const long n_wqkv = (long)EMB * E3;
    const long n_wout = (long)EMB * EMB;
    const long n_x    = (long)MROWS * EMB;

    minmax_partial_kernel<<<64, 256, 0, stream>>>(w_qkv, n_wqkv, partA);
    minmax_final_kernel<<<1, 256, 0, stream>>>(partA, 64, paramsQ);
    minmax_partial_kernel<<<64, 256, 0, stream>>>(w_out, n_wout, partB);
    minmax_final_kernel<<<1, 256, 0, stream>>>(partB, 64, paramsO);

    quantize_bf16_kernel<<<2048, 256, 0, stream>>>(w_qkv, wqkv_bf, n_wqkv, paramsQ);
    quantize_bf16_kernel<<<1024, 256, 0, stream>>>(w_out, wout_bf, n_wout, paramsO);
    cvt_bf16_kernel<<<2048, 256, 0, stream>>>(x, x_bf, n_x);

    gemm_bf16_kernel<true><<<dim3(E3 / 128, MROWS / 128), 256, 0, stream>>>(
        x_bf, wqkv_bf, (void*)qkv_bf, MROWS, E3, EMB);

    {
        long total = (long)BATCH * SEQ * 2 * HEADS * (HD / 2);
        int blocks = (int)((total + 255) / 256);
        rope_kernel<<<blocks, 256, 0, stream>>>(qkv_bf);
    }

    flash_attn_kernel<<<dim3(SEQ / 128, BATCH * HEADS), 256, 0, stream>>>(
        qkv_bf, attn_bf);

    gemm_bf16_kernel<false><<<dim3(EMB / 128, MROWS / 128), 256, 0, stream>>>(
        attn_bf, wout_bf, (void*)out, MROWS, EMB, EMB);
}
